// AttentionLayer_22617297781200
// MI455X (gfx1250) — matrix operand order repaired and checked by reading, not compiled, not run on hardware
//
#include <hip/hip_runtime.h>
#include <hip/hip_bf16.h>
#include <math.h>

// ---------------------------------------------------------------------------
// Types for CDNA5 WMMA (wave32): v_wmma_f32_16x16x32_bf16
// ---------------------------------------------------------------------------
typedef __attribute__((ext_vector_type(16))) __bf16 v16bf;
typedef __attribute__((ext_vector_type(8)))  __bf16 v8bf;
typedef __attribute__((ext_vector_type(4)))  __bf16 v4bf;
typedef __attribute__((ext_vector_type(8)))  float  v8f;

static __device__ __forceinline__ v16bf join8(v8bf lo, v8bf hi) {
  v16bf r;
#pragma unroll
  for (int i = 0; i < 8; ++i) { r[i] = lo[i]; r[i + 8] = hi[i]; }
  return r;
}

// load 8 consecutive fp32 and convert to 8 bf16 (for on-the-fly A fragments)
static __device__ __forceinline__ v8bf cvt8(const float* __restrict__ p) {
  float4 a = *(const float4*)p;
  float4 b = *(const float4*)(p + 4);
  v8bf r;
  r[0] = (__bf16)a.x; r[1] = (__bf16)a.y; r[2] = (__bf16)a.z; r[3] = (__bf16)a.w;
  r[4] = (__bf16)b.x; r[5] = (__bf16)b.y; r[6] = (__bf16)b.z; r[7] = (__bf16)b.w;
  return r;
}

#define WMMA_BF16(a, b, c) \
  __builtin_amdgcn_wmma_f32_16x16x32_bf16(false, (a), false, (b), (short)0, (c), false, false)

// Problem constants: B=16, T=12, N=325, D=128, H=8, hd=16, F=256
#define SROWS   62400      // B*T*N
#define DDIM    128
#define FDIM    256
#define NNODES  325
#define NQT     21         // ceil(325/16)
#define NPAD    336        // 21*16
#define APAD    352        // 11*32
#define NHEADS  8

// ---------------------------------------------------------------------------
// Pack a fp32 weight W[K][N] (row-major) into WMMA B-fragment lane order:
//   P[(((kt*(N/16)+nt)*32 + lane)*16 + i)] = (bf16) W[kt*32 + (lane>>4)*16 + i][nt*16 + (lane&15)]
// ---------------------------------------------------------------------------
__global__ void pack_b_kernel(const float* __restrict__ W, __bf16* __restrict__ P,
                              int K, int N) {
  int idx = blockIdx.x * blockDim.x + threadIdx.x;
  if (idx >= K * N) return;
  int i    = idx & 15;
  int lane = (idx >> 4) & 31;
  int f    = idx >> 9;
  int ntn  = N >> 4;
  int nt   = f % ntn;
  int kt   = f / ntn;
  int kk   = kt * 32 + (lane >> 4) * 16 + i;
  int n    = nt * 16 + (lane & 15);
  P[idx] = (__bf16)W[(size_t)kk * N + n];
}

// ---------------------------------------------------------------------------
// Fused QKV: reads fp32 x ONCE, converts to bf16 fragments in-register,
// accumulates all three projections per A load.  Wave -> 16x64 strip x3.
// ---------------------------------------------------------------------------
__global__ __launch_bounds__(256) void qkv_kernel(
    const float* __restrict__ X,
    const __bf16* __restrict__ Pq, const __bf16* __restrict__ Pk,
    const __bf16* __restrict__ Pv,
    const float* __restrict__ bq, const float* __restrict__ bk,
    const float* __restrict__ bv,
    __bf16* __restrict__ Oq, __bf16* __restrict__ Ok, __bf16* __restrict__ Ov,
    int M) {
  int gtid = blockIdx.x * 256 + threadIdx.x;
  int wid  = gtid >> 5;
  int totalWaves = (M >> 4) * 2;            // N=128 -> 2 column groups
  if (wid >= totalWaves) return;
  int lane = threadIdx.x & 31;
  int mt = wid >> 1;
  int ng = wid & 1;
  int cl = lane & 15, hh = lane >> 4;

  const __bf16* P[3]   = {Pq, Pk, Pv};
  const float*  bias[3] = {bq, bk, bv};
  __bf16*       O[3]   = {Oq, Ok, Ov};

  v8f c[3][4];
#pragma unroll
  for (int w = 0; w < 3; ++w)
#pragma unroll
    for (int j = 0; j < 4; ++j) c[w][j] = {};

  const float* Abase = X + (size_t)(mt * 16 + cl) * DDIM + hh * 16;
#pragma unroll
  for (int kt = 0; kt < 4; ++kt) {          // K = 128
    v16bf a = join8(cvt8(Abase + kt * 32), cvt8(Abase + kt * 32 + 8));
#pragma unroll
    for (int w = 0; w < 3; ++w) {
#pragma unroll
      for (int j = 0; j < 4; ++j) {
        int nt = ng * 4 + j;
        const __bf16* bp = P[w] + ((((size_t)kt * 8 + nt) * 32 + lane) << 4);
        v16bf b = join8(*(const v8bf*)bp, *(const v8bf*)(bp + 8));
        c[w][j] = WMMA_BF16(a, b, c[w][j]);
      }
    }
  }
#pragma unroll
  for (int w = 0; w < 3; ++w) {
#pragma unroll
    for (int j = 0; j < 4; ++j) {
      int col = ng * 64 + j * 16 + cl;
      float bvv = bias[w][col];
#pragma unroll
      for (int r = 0; r < 8; ++r) {
        int row = mt * 16 + r + 8 * hh;
        O[w][(size_t)row * DDIM + col] = (__bf16)(c[w][j][r] + bvv);
      }
    }
  }
}

// ---------------------------------------------------------------------------
// Generic bf16 WMMA GEMM (used for FFN1 with GELU): wave -> 16x64 strip.
// ---------------------------------------------------------------------------
template <bool SF32, bool SB16, bool DOGELU>
__global__ __launch_bounds__(256) void gemm_bf16_kernel(
    const __bf16* __restrict__ A, const __bf16* __restrict__ Bp,
    const float* __restrict__ bias, float* __restrict__ Cf,
    __bf16* __restrict__ Cb, int M, int N, int K) {
  int gtid = blockIdx.x * 256 + threadIdx.x;
  int wid  = gtid >> 5;
  int lane = threadIdx.x & 31;
  int ngroups    = N >> 6;
  int totalWaves = (M >> 4) * ngroups;
  if (wid >= totalWaves) return;
  int mt = wid / ngroups;
  int ng = wid % ngroups;
  int cl = lane & 15, hh = lane >> 4;

  v8f c[4];
#pragma unroll
  for (int j = 0; j < 4; ++j) c[j] = {};

  int ksteps = K >> 5;
  int ntn    = N >> 4;
  const __bf16* Abase = A + (size_t)(mt * 16 + cl) * K + hh * 16;
  for (int kt = 0; kt < ksteps; ++kt) {
    const __bf16* ap = Abase + kt * 32;
    v16bf a = join8(*(const v8bf*)ap, *(const v8bf*)(ap + 8));
#pragma unroll
    for (int j = 0; j < 4; ++j) {
      int nt = ng * 4 + j;
      const __bf16* bp = Bp + ((((size_t)kt * ntn + nt) * 32 + lane) << 4);
      v16bf b = join8(*(const v8bf*)bp, *(const v8bf*)(bp + 8));
      c[j] = WMMA_BF16(a, b, c[j]);
    }
  }
#pragma unroll
  for (int j = 0; j < 4; ++j) {
    int col = ng * 64 + j * 16 + cl;
    float bv = bias ? bias[col] : 0.0f;
#pragma unroll
    for (int r = 0; r < 8; ++r) {
      int row = mt * 16 + r + 8 * hh;
      float v = c[j][r] + bv;
      if (DOGELU) v = 0.5f * v * (1.0f + erff(v * 0.70710678118654752f));
      size_t off = (size_t)row * N + col;
      if (SF32) Cf[off] = v;
      if (SB16) Cb[off] = (__bf16)v;
    }
  }
}

// ---------------------------------------------------------------------------
// GEMM (N=128) with fused bias + residual + LayerNorm epilogue.
// A wave computes a full 16x128 row strip, so each output row lives entirely
// in one 16-lane half (row = r + 8*hh, cols = j*16 + (lane&15)).  Row mean/var
// reduce with shfl_xor masks 1,2,4,8 (never crossing the half boundary).
//   out = LN(X + A@W + bias) * g + b   -> fp32 (outF) and/or bf16 (outB)
// ---------------------------------------------------------------------------
__global__ __launch_bounds__(256) void gemm_ln_kernel(
    const __bf16* __restrict__ A, const __bf16* __restrict__ Bp,
    const float* __restrict__ bias, const float* __restrict__ X,
    const float* __restrict__ g, const float* __restrict__ b,
    float* __restrict__ outF, __bf16* __restrict__ outB, int M, int K) {
  int gtid = blockIdx.x * 256 + threadIdx.x;
  int wid  = gtid >> 5;
  if (wid >= (M >> 4)) return;
  int lane = threadIdx.x & 31;
  int mt = wid;
  int cl = lane & 15, hh = lane >> 4;

  v8f c[8];
#pragma unroll
  for (int j = 0; j < 8; ++j) c[j] = {};

  int ksteps = K >> 5;
  const __bf16* Abase = A + (size_t)(mt * 16 + cl) * K + hh * 16;
  for (int kt = 0; kt < ksteps; ++kt) {
    const __bf16* ap = Abase + kt * 32;
    v16bf a = join8(*(const v8bf*)ap, *(const v8bf*)(ap + 8));
#pragma unroll
    for (int j = 0; j < 8; ++j) {
      const __bf16* bp = Bp + ((((size_t)kt * 8 + j) * 32 + lane) << 4);
      v16bf bb = join8(*(const v8bf*)bp, *(const v8bf*)(bp + 8));
      c[j] = WMMA_BF16(a, bb, c[j]);
    }
  }

  // epilogue: bias + residual, per-row LN stats, normalize, store
  float val[8][8];
  float gv[8], bv[8];
#pragma unroll
  for (int j = 0; j < 8; ++j) {
    int col = j * 16 + cl;
    float bia = bias[col];
    gv[j] = g[col];
    bv[j] = b[col];
#pragma unroll
    for (int r = 0; r < 8; ++r) {
      int row = mt * 16 + r + 8 * hh;
      val[j][r] = c[j][r] + bia + X[(size_t)row * DDIM + col];
    }
  }
#pragma unroll
  for (int r = 0; r < 8; ++r) {
    float s = 0.0f, sq = 0.0f;
#pragma unroll
    for (int j = 0; j < 8; ++j) { float v = val[j][r]; s += v; sq += v * v; }
#pragma unroll
    for (int m = 1; m < 16; m <<= 1) {      // stays within 16-lane half
      s  += __shfl_xor(s, m, 32);
      sq += __shfl_xor(sq, m, 32);
    }
    float mean = s * (1.0f / 128.0f);
    float var  = sq * (1.0f / 128.0f) - mean * mean;
    float rstd = rsqrtf(var + 1e-5f);
    int row = mt * 16 + r + 8 * hh;
#pragma unroll
    for (int j = 0; j < 8; ++j) {
      int col = j * 16 + cl;
      float o = (val[j][r] - mean) * rstd * gv[j] + bv[j];
      size_t off = (size_t)row * DDIM + col;
      if (outF) outF[off] = o;
      if (outB) outB[off] = (__bf16)o;
    }
  }
}

// ---------------------------------------------------------------------------
// Attention: one block (4 waves) per (b,t,h).  scores = (Q/4) K^T, softmax,
// out = P V, all matmuls on WMMA.  Qs/Ks rows are 32 K-values wide with the
// upper 16 zeroed at staging so all fragment loads are unconditional.
// ---------------------------------------------------------------------------
__global__ __launch_bounds__(128) void attn_kernel(
    const __bf16* __restrict__ qg, const __bf16* __restrict__ kg,
    const __bf16* __restrict__ vg, __bf16* __restrict__ og) {
  __shared__ __bf16 Qs[NPAD * 32];        // [node][K=32], K>=16 zero, pre-scaled 0.25
  __shared__ __bf16 Ks[NPAD * 32];        // [node][K=32], K>=16 zero
  __shared__ __bf16 Vt[16 * NPAD];        // [hd][node] transposed
  __shared__ float  Sc[4][16 * NPAD];     // per-wave scores
  __shared__ __bf16 Pb[4][16 * APAD];     // per-wave probs

  int bh  = blockIdx.x;
  int h   = bh % NHEADS;
  int bt  = bh / NHEADS;
  int tid = threadIdx.x;
  int wid = tid >> 5;
  int lane = tid & 31;
  size_t base = (size_t)bt * NNODES * DDIM + (size_t)h * 16;

  __bf16 bz = (__bf16)0.0f;
  v8bf zv;
#pragma unroll
  for (int d = 0; d < 8; ++d) zv[d] = bz;

  for (int r = tid; r < NPAD; r += 128) {
    v8bf q0 = zv, q1 = zv, k0 = zv, k1 = zv, w0 = zv, w1 = zv;
    if (r < NNODES) {
      const v8bf* qp = (const v8bf*)(qg + base + (size_t)r * DDIM);
      const v8bf* kp = (const v8bf*)(kg + base + (size_t)r * DDIM);
      const v8bf* vp = (const v8bf*)(vg + base + (size_t)r * DDIM);
      q0 = qp[0]; q1 = qp[1];
#pragma unroll
      for (int d = 0; d < 8; ++d) {
        q0[d] = (__bf16)((float)q0[d] * 0.25f);
        q1[d] = (__bf16)((float)q1[d] * 0.25f);
      }
      k0 = kp[0]; k1 = kp[1];
      w0 = vp[0]; w1 = vp[1];
    }
    v8bf* qrow = (v8bf*)(Qs + r * 32);
    qrow[0] = q0; qrow[1] = q1; qrow[2] = zv; qrow[3] = zv;
    v8bf* krow = (v8bf*)(Ks + r * 32);
    krow[0] = k0; krow[1] = k1; krow[2] = zv; krow[3] = zv;
#pragma unroll
    for (int d = 0; d < 8; ++d) {
      Vt[d * NPAD + r]       = w0[d];
      Vt[(d + 8) * NPAD + r] = w1[d];
    }
  }
  __syncthreads();

  int col = lane & 15;
  int hh  = lane >> 4;

  for (int qt0 = 0; qt0 < 24; qt0 += 4) {
    int qt = qt0 + wid;
    bool act = (qt < NQT);

    if (act) {
      const __bf16* ap = Qs + (qt * 16 + col) * 32 + hh * 16;
      v16bf a = join8(*(const v8bf*)ap, *(const v8bf*)(ap + 8));
      for (int nt = 0; nt < NQT; ++nt) {
        const __bf16* bp = Ks + (nt * 16 + col) * 32 + hh * 16;
        v16bf b = join8(*(const v8bf*)bp, *(const v8bf*)(bp + 8));
        v8f c = {};
        c = WMMA_BF16(a, b, c);
#pragma unroll
        for (int r = 0; r < 8; ++r)
          Sc[wid][(r + 8 * hh) * NPAD + nt * 16 + col] = c[r];
      }
    }
    __syncthreads();

    if (act) {
      float* srow  = &Sc[wid][(lane & 15) * NPAD];
      __bf16* prow = &Pb[wid][(lane & 15) * APAD];
      float mx = -3.0e38f;
      for (int c4 = hh; c4 < 84; c4 += 2) {
        float4 v = *(const float4*)(srow + c4 * 4);
        int c0 = c4 * 4;
        float m0 = (c0 + 0 < NNODES) ? v.x : -3.0e38f;
        float m1 = (c0 + 1 < NNODES) ? v.y : -3.0e38f;
        float m2 = (c0 + 2 < NNODES) ? v.z : -3.0e38f;
        float m3 = (c0 + 3 < NNODES) ? v.w : -3.0e38f;
        mx = fmaxf(mx, fmaxf(fmaxf(m0, m1), fmaxf(m2, m3)));
      }
      mx = fmaxf(mx, __shfl_xor(mx, 16, 32));
      float sum = 0.0f;
      for (int c4 = hh; c4 < 84; c4 += 2) {
        float4 v = *(const float4*)(srow + c4 * 4);
        int c0 = c4 * 4;
        float e0 = (c0 + 0 < NNODES) ? __expf(v.x - mx) : 0.0f;
        float e1 = (c0 + 1 < NNODES) ? __expf(v.y - mx) : 0.0f;
        float e2 = (c0 + 2 < NNODES) ? __expf(v.z - mx) : 0.0f;
        float e3 = (c0 + 3 < NNODES) ? __expf(v.w - mx) : 0.0f;
        sum += e0 + e1 + e2 + e3;
        float4 ev = {e0, e1, e2, e3};
        *(float4*)(srow + c4 * 4) = ev;
      }
      sum += __shfl_xor(sum, 16, 32);
      float inv = 1.0f / sum;
      for (int c4 = hh; c4 < 88; c4 += 2) {
        v4bf p;
        if (c4 < 84) {
          float4 v = *(const float4*)(srow + c4 * 4);
          p[0] = (__bf16)(v.x * inv);
          p[1] = (__bf16)(v.y * inv);
          p[2] = (__bf16)(v.z * inv);
          p[3] = (__bf16)(v.w * inv);
        } else {
          p[0] = bz; p[1] = bz; p[2] = bz; p[3] = bz;
        }
        *(v4bf*)(prow + c4 * 4) = p;
      }
    }
    __syncthreads();

    if (act) {
      v8f c2 = {};
      for (int kt = 0; kt < APAD / 32; ++kt) {
        const __bf16* ap = &Pb[wid][(lane & 15) * APAD + kt * 32 + hh * 16];
        v16bf a = join8(*(const v8bf*)ap, *(const v8bf*)(ap + 8));
        const __bf16* bp = Vt + col * NPAD + kt * 32 + hh * 16;
        v16bf b = join8(*(const v8bf*)bp, *(const v8bf*)(bp + 8));
        c2 = WMMA_BF16(a, b, c2);
      }
#pragma unroll
      for (int r = 0; r < 8; ++r) {
        int n = qt * 16 + r + 8 * hh;
        if (n < NNODES) og[base + (size_t)n * DDIM + col] = (__bf16)c2[r];
      }
    }
    __syncthreads();
  }
}

// ---------------------------------------------------------------------------
// Host-side orchestration
// ---------------------------------------------------------------------------
extern "C" void kernel_launch(void* const* d_in, const int* in_sizes, int n_in,
                              void* d_out, int out_size, void* d_ws, size_t ws_size,
                              hipStream_t stream) {
  const float* x     = (const float*)d_in[0];
  const float* Wq    = (const float*)d_in[1];
  const float* bq    = (const float*)d_in[2];
  const float* Wk    = (const float*)d_in[3];
  const float* bk    = (const float*)d_in[4];
  const float* Wv    = (const float*)d_in[5];
  const float* bv    = (const float*)d_in[6];
  const float* Wo    = (const float*)d_in[7];
  const float* bo    = (const float*)d_in[8];
  const float* ln1g  = (const float*)d_in[9];
  const float* ln1b  = (const float*)d_in[10];
  const float* W1    = (const float*)d_in[11];
  const float* b1    = (const float*)d_in[12];
  const float* W2    = (const float*)d_in[13];
  const float* b2    = (const float*)d_in[14];
  const float* ln2g  = (const float*)d_in[15];
  const float* ln2b  = (const float*)d_in[16];

  char* ws = (char*)d_ws;
  const size_t R16 = (size_t)16 << 20;   // 16 MiB region unit
  // Region plan (total = 7*16MiB + 256KiB ~= 112.25 MiB):
  __bf16* qb    = (__bf16*)(ws + 0 * R16);  // q bf16        [0,16)
  __bf16* kb    = (__bf16*)(ws + 1 * R16);  // k bf16        [16,32)
  __bf16* vb    = (__bf16*)(ws + 2 * R16);  // v bf16        [32,48)
  __bf16* attnb = (__bf16*)(ws + 3 * R16);  // attn out bf16 [48,64)
  float*  hbuf  = (float*) (ws + 4 * R16);  // LN1 out fp32  [64,96)
  __bf16* hb    = (__bf16*)(ws + 6 * R16);  // LN1 out bf16  [96,112)
  __bf16* f1    = (__bf16*)(ws + 0 * R16);  // FFN1 out bf16 [0,32) (reuses qb,kb)
  __bf16* pW    = (__bf16*)(ws + 7 * R16);  // packed weights (256 KiB)
  __bf16* pWq = pW;
  __bf16* pWk = pW + 16384;
  __bf16* pWv = pW + 32768;
  __bf16* pWo = pW + 49152;
  __bf16* pW1 = pW + 65536;                 // 128x256
  __bf16* pW2 = pW + 98304;                 // 256x128

  const int S = SROWS;                      // 62400 rows

  // 1) pack weights into WMMA B-fragment order
  pack_b_kernel<<<64,  256, 0, stream>>>(Wq, pWq, 128, 128);
  pack_b_kernel<<<64,  256, 0, stream>>>(Wk, pWk, 128, 128);
  pack_b_kernel<<<64,  256, 0, stream>>>(Wv, pWv, 128, 128);
  pack_b_kernel<<<64,  256, 0, stream>>>(Wo, pWo, 128, 128);
  pack_b_kernel<<<128, 256, 0, stream>>>(W1, pW1, 128, 256);
  pack_b_kernel<<<128, 256, 0, stream>>>(W2, pW2, 256, 128);

  // 2) fused QKV (reads fp32 x once): waves = 3900*2 = 7800 -> 975 blocks
  qkv_kernel<<<975, 256, 0, stream>>>(x, pWq, pWk, pWv, bq, bk, bv, qb, kb, vb, S);

  // 3) attention: one block per (b,t,h) = 1536
  attn_kernel<<<1536, 128, 0, stream>>>(qb, kb, vb, attnb);

  // 4) Wo projection fused with residual + LN1:  h = LN(x + attn@Wo + bo)
  //    waves = 3900 -> 488 blocks
  gemm_ln_kernel<<<488, 256, 0, stream>>>(attnb, pWo, bo, x, ln1g, ln1b,
                                          hbuf, hb, S, 128);

  // 5) FFN1: GELU(h @ W1 + b1) bf16   waves = 3900*4 = 15600 -> 1950 blocks
  gemm_bf16_kernel<false, true, true><<<1950, 256, 0, stream>>>(
      hb, pW1, b1, nullptr, f1, S, 256, 128);

  // 6) FFN2 fused with residual + LN2:  out = LN(h + f1@W2 + b2) -> d_out fp32
  gemm_ln_kernel<<<488, 256, 0, stream>>>(f1, pW2, b2, hbuf, ln2g, ln2b,
                                          (float*)d_out, nullptr, S, 256);
}
